// SingleHeadAttention_34205119546036
// MI455X (gfx1250) — matrix operand order repaired and checked by reading, not compiled, not run on hardware
//
#include <hip/hip_runtime.h>
#include <hip/hip_bf16.h>

typedef __bf16 bf16_t;
typedef __attribute__((ext_vector_type(16))) __bf16 v16bf;
typedef __attribute__((ext_vector_type(8)))  float  v8f;

__device__ inline v8f wmma_bf16(v16bf a, v16bf b, v8f c) {
    return __builtin_amdgcn_wmma_f32_16x16x32_bf16(
        /*neg_a=*/false, a, /*neg_b=*/false, b,
        /*c_mod=*/(short)0, c, /*reuse_a=*/false, /*reuse_b=*/false);
}

// A fragment (16x32 bf16, MxK): tile row-major [m][k], stride lda elements.
// Lane L holds row m=L%16; elements 0..7 -> K = k0 + (L/16)*8 + i,
// elements 8..15 -> K = k0 + 16 + (L/16)*8 + (i-8).  (ISA 05_wmma 16-bit A layout)
__device__ inline v16bf frag_a(const bf16_t* tile, int lda, int m, int k0, int half) {
    const bf16_t* p0 = tile + m * lda + k0 + half * 8;
    const bf16_t* p1 = p0 + 16;
    v16bf a;
#pragma unroll
    for (int i = 0; i < 8; ++i) { a[i] = p0[i]; a[8 + i] = p1[i]; }
    return a;
}

// B fragment (32x16 bf16, KxN) from a K-major transposed tile Bt[n][k], stride ldk.
// Lane L holds col n=L%16; elements 0..7 -> K = k0 + (L/16)*8 + i,
// elements 8..15 -> K = k0 + 16 + (L/16)*8 + (i-8) (same as A).
__device__ inline v16bf frag_b(const bf16_t* btile, int ldk, int n, int k0, int half) {
    const bf16_t* p0 = btile + n * ldk + k0 + half * 8;
    const bf16_t* p1 = p0 + 16;
    v16bf b;
#pragma unroll
    for (int i = 0; i < 8; ++i) { b[i] = p0[i]; b[8 + i] = p1[i]; }
    return b;
}

// ---------------------------------------------------------------- fp32 -> bf16
__global__ __launch_bounds__(256) void cvt_bf16_kernel(const float* __restrict__ in,
                                                       bf16_t* __restrict__ out, int n) {
    int i = (blockIdx.x * 256 + threadIdx.x) * 4;
    if (i < n) {
        float4 v = *(const float4*)&in[i];
        bf16_t o[4];
        o[0] = (bf16_t)v.x; o[1] = (bf16_t)v.y; o[2] = (bf16_t)v.z; o[3] = (bf16_t)v.w;
        *(uint2*)&out[i] = *(uint2*)o;
    }
}

// ------------------------------------------------- QKV projection GEMM (+bias)
// C(bf16) = X(bf16, MxK row-major) * W(bf16, KxN row-major) + bias(f32)
// Double-buffered LDS: next K-tile's global loads overlap current tile's WMMAs.
#define GM_BM 128
#define GM_BN 128
#define GM_BK 64

__global__ __launch_bounds__(256) void qkv_gemm_kernel(const bf16_t* __restrict__ X,
                                                       const bf16_t* __restrict__ W,
                                                       const float*  __restrict__ bias,
                                                       bf16_t* __restrict__ out,
                                                       int M, int K, int N) {
    __shared__ bf16_t As[2][GM_BM][GM_BK];   // 2 x 16 KB
    __shared__ bf16_t Bt[2][GM_BN][GM_BK];   // 2 x 16 KB (W tile transposed, k contiguous)

    const int tid  = threadIdx.x;
    const int lane = tid & 31;
    const int wave = tid >> 5;            // 8 waves
    const int wm   = wave & 1;            // 2 waves in M
    const int wn   = wave >> 1;           // 4 waves in N
    const int half = lane >> 4;
    const int lm   = lane & 15;
    const int bm0  = blockIdx.y * GM_BM;
    const int bn0  = blockIdx.x * GM_BN;

    auto load_tile = [&](int iter, int buf) {
        int k0 = iter * GM_BK;
        // A tile: 128x64 bf16 = 8192 elems; 256 threads x 4 x uint4(8 elems)
#pragma unroll
        for (int c = 0; c < 4; ++c) {
            int e = (tid + c * 256) * 8;
            int r = e >> 6, col = e & 63;
            *(uint4*)&As[buf][r][col] = *(const uint4*)&X[(size_t)(bm0 + r) * K + k0 + col];
        }
        // B tile transposed: W[k0+kk][bn0+nn] -> Bt[nn][kk]
#pragma unroll
        for (int c = 0; c < 4; ++c) {
            int e  = (tid + c * 256) * 8;
            int kk = e >> 7, nn = e & 127;
            uint4 v = *(const uint4*)&W[(size_t)(k0 + kk) * N + bn0 + nn];
            bf16_t tmp[8];
            *(uint4*)tmp = v;
#pragma unroll
            for (int j = 0; j < 8; ++j) Bt[buf][nn + j][kk] = tmp[j];
        }
    };

    v8f acc[4][2];                        // 64x32 wave tile = 4x2 fragments
#pragma unroll
    for (int mt = 0; mt < 4; ++mt)
#pragma unroll
        for (int nt = 0; nt < 2; ++nt) acc[mt][nt] = (v8f){};

    const int niter = K / GM_BK;
    load_tile(0, 0);
    __syncthreads();

    for (int it = 0; it < niter; ++it) {
        int buf = it & 1;
        if (it + 1 < niter) load_tile(it + 1, buf ^ 1);   // overlaps with WMMAs below

#pragma unroll
        for (int ks = 0; ks < GM_BK; ks += 32) {
            v16bf bfr[2];
#pragma unroll
            for (int nt = 0; nt < 2; ++nt)
                bfr[nt] = frag_b(&Bt[buf][0][0], GM_BK, wn * 32 + nt * 16 + lm, ks, half);
#pragma unroll
            for (int mt = 0; mt < 4; ++mt) {
                v16bf afr = frag_a(&As[buf][0][0], GM_BK, wm * 64 + mt * 16 + lm, ks, half);
#pragma unroll
                for (int nt = 0; nt < 2; ++nt)
                    acc[mt][nt] = wmma_bf16(afr, bfr[nt], acc[mt][nt]);
            }
        }
        __syncthreads();   // next-tile LDS writes visible; buf free to overwrite next round
    }

    // epilogue: + bias, convert, store
#pragma unroll
    for (int mt = 0; mt < 4; ++mt)
#pragma unroll
        for (int nt = 0; nt < 2; ++nt) {
            int n = bn0 + wn * 32 + nt * 16 + lm;
            float bv = bias[n];
#pragma unroll
            for (int r = 0; r < 8; ++r) {
                int m = bm0 + wm * 64 + mt * 16 + half * 8 + r;
                out[(size_t)m * N + n] = (bf16_t)(acc[mt][nt][r] + bv);
            }
        }
}

// -------------------------------------------------------- flash attention
#define AT_BM 32
#define AT_BN 32
#define AT_H  1024

__global__ __launch_bounds__(256) void attn_kernel(const bf16_t* __restrict__ Q,
                                                   const bf16_t* __restrict__ K,
                                                   const bf16_t* __restrict__ V,
                                                   float* __restrict__ O, int T) {
    __shared__ bf16_t Qs[AT_BM][AT_H];    // 64 KB
    __shared__ bf16_t Ks[AT_BN][AT_H];    // 64 KB
    __shared__ bf16_t Vt[AT_H][AT_BN];    // 64 KB (transposed: [h][kv])
    __shared__ float  Ss[AT_BM][AT_BN];   // 4 KB
    __shared__ bf16_t Ps[AT_BM][AT_BN];   // 2 KB
    __shared__ float  mRow[AT_BM], lRow[AT_BM], aRow[AT_BM];

    const int tid  = threadIdx.x;
    const int lane = tid & 31;
    const int wave = tid >> 5;            // 8 waves: wave owns h-slice [wave*128, +128)
    const int half = lane >> 4;
    const int lm   = lane & 15;
    const int b    = blockIdx.y;
    const int mb   = blockIdx.x;
    const size_t base = (size_t)b * T * AT_H;
    const float scale = 0.03125f;         // 1/sqrt(1024)

    // Q tile: 32x1024 bf16 = 64 KB -> 256 threads x 16 x uint4
#pragma unroll
    for (int c = 0; c < 16; ++c) {
        int e = (tid + c * 256) * 8;
        int r = e >> 10, col = e & 1023;
        *(uint4*)&Qs[r][col] = *(const uint4*)&Q[base + (size_t)(mb * AT_BM + r) * AT_H + col];
    }
    if (tid < AT_BM) { mRow[tid] = -INFINITY; lRow[tid] = 0.f; }

    v8f acc[2][8];                        // O slice 32x128 f32 = 128 VGPRs
#pragma unroll
    for (int mt = 0; mt < 2; ++mt)
#pragma unroll
        for (int nt = 0; nt < 8; ++nt) acc[mt][nt] = (v8f){};

    for (int kb = 0; kb <= mb; ++kb) {
        __syncthreads();                  // previous iter done with Ks/Vt/Ps
        // K tile
#pragma unroll
        for (int c = 0; c < 16; ++c) {
            int e = (tid + c * 256) * 8;
            int r = e >> 10, col = e & 1023;
            *(uint4*)&Ks[r][col] = *(const uint4*)&K[base + (size_t)(kb * AT_BN + r) * AT_H + col];
        }
        // V tile, transposed into LDS
#pragma unroll
        for (int c = 0; c < 16; ++c) {
            int e = (tid + c * 256) * 8;
            int r = e >> 10, col = e & 1023;
            uint4 v = *(const uint4*)&V[base + (size_t)(kb * AT_BN + r) * AT_H + col];
            bf16_t tmp[8];
            *(uint4*)tmp = v;
#pragma unroll
            for (int j = 0; j < 8; ++j) Vt[col + j][r] = tmp[j];
        }
        // zero S
        {
            float* ss = &Ss[0][0];
#pragma unroll
            for (int c = 0; c < 4; ++c) ss[tid + c * 256] = 0.f;
        }
        // warm L2 for the next KV block while this block computes
        if (kb + 1 <= mb) {
            const bf16_t* nK = &K[base + (size_t)((kb + 1) * AT_BN) * AT_H];
            const bf16_t* nV = &V[base + (size_t)((kb + 1) * AT_BN) * AT_H];
#pragma unroll
            for (int c = 0; c < 2; ++c) {
                __builtin_prefetch(nK + (size_t)(tid + c * 256) * 64, 0, 3);
                __builtin_prefetch(nV + (size_t)(tid + c * 256) * 64, 0, 3);
            }
        }
        __syncthreads();

        // partial S = Q * K^T over this wave's k-slice of 128
        v8f sAcc[2][2];
#pragma unroll
        for (int mt = 0; mt < 2; ++mt)
#pragma unroll
            for (int nt = 0; nt < 2; ++nt) sAcc[mt][nt] = (v8f){};
#pragma unroll
        for (int ks = 0; ks < 128; ks += 32) {
            int k0 = wave * 128 + ks;
            v16bf bfr[2];
#pragma unroll
            for (int nt = 0; nt < 2; ++nt)
                bfr[nt] = frag_b(&Ks[0][0], AT_H, nt * 16 + lm, k0, half);
#pragma unroll
            for (int mt = 0; mt < 2; ++mt) {
                v16bf afr = frag_a(&Qs[0][0], AT_H, mt * 16 + lm, k0, half);
#pragma unroll
                for (int nt = 0; nt < 2; ++nt)
                    sAcc[mt][nt] = wmma_bf16(afr, bfr[nt], sAcc[mt][nt]);
            }
        }
        // reduce partials across waves
#pragma unroll
        for (int mt = 0; mt < 2; ++mt)
#pragma unroll
            for (int nt = 0; nt < 2; ++nt)
#pragma unroll
                for (int r = 0; r < 8; ++r)
                    atomicAdd(&Ss[mt * 16 + half * 8 + r][nt * 16 + lm], sAcc[mt][nt][r]);
        __syncthreads();

        // online softmax: all 8 waves, 4 rows per wave, 8 lanes per row, 4 cols per lane
        {
            int r  = wave * 4 + (lane >> 3);          // row 0..31
            int cg = lane & 7;                        // column group
            int nvalid = (kb < mb) ? AT_BN : (r + 1); // causal
            float mOld = mRow[r];
            float mNew = mOld;
            float sv[4];
#pragma unroll
            for (int j = 0; j < 4; ++j) {
                int c = cg + j * 8;
                sv[j] = (c < nvalid) ? Ss[r][c] * scale : -INFINITY;
                mNew = fmaxf(mNew, sv[j]);
            }
#pragma unroll
            for (int off = 1; off < 8; off <<= 1)
                mNew = fmaxf(mNew, __shfl_xor(mNew, off, 32));
            float alpha = __expf(mOld - mNew);
            float psum = 0.f;
#pragma unroll
            for (int j = 0; j < 4; ++j) {
                int c = cg + j * 8;
                float p = __expf(sv[j] - mNew);       // masked sv = -inf -> p = 0
                Ps[r][c] = (bf16_t)p;
                psum += p;
            }
#pragma unroll
            for (int off = 1; off < 8; off <<= 1)
                psum += __shfl_xor(psum, off, 32);
            if (cg == 0) {
                mRow[r] = mNew;
                lRow[r] = lRow[r] * alpha + psum;
                aRow[r] = alpha;
            }
        }
        __syncthreads();

        // rescale O accumulators, then O += P * V
#pragma unroll
        for (int mt = 0; mt < 2; ++mt) {
            float al[8];
#pragma unroll
            for (int r = 0; r < 8; ++r) al[r] = aRow[mt * 16 + half * 8 + r];
#pragma unroll
            for (int nt = 0; nt < 8; ++nt)
#pragma unroll
                for (int r = 0; r < 8; ++r) acc[mt][nt][r] *= al[r];
        }
#pragma unroll
        for (int mt = 0; mt < 2; ++mt) {
            v16bf pfr = frag_a(&Ps[0][0], AT_BN, mt * 16 + lm, 0, half);
#pragma unroll
            for (int nt = 0; nt < 8; ++nt) {
                v16bf vfr = frag_b(&Vt[0][0], AT_BN, wave * 128 + nt * 16 + lm, 0, half);
                acc[mt][nt] = wmma_bf16(pfr, vfr, acc[mt][nt]);
            }
        }
    }
    __syncthreads();

    // O = acc / l, store fp32
#pragma unroll
    for (int mt = 0; mt < 2; ++mt) {
        float li[8];
#pragma unroll
        for (int r = 0; r < 8; ++r) li[r] = 1.f / lRow[mt * 16 + half * 8 + r];
#pragma unroll
        for (int nt = 0; nt < 8; ++nt) {
            int h = wave * 128 + nt * 16 + lm;
#pragma unroll
            for (int r = 0; r < 8; ++r) {
                int row = mb * AT_BM + mt * 16 + half * 8 + r;
                O[base + (size_t)row * AT_H + h] = acc[mt][nt][r] * li[r];
            }
        }
    }
}

extern "C" void kernel_launch(void* const* d_in, const int* in_sizes, int n_in,
                              void* d_out, int out_size, void* d_ws, size_t ws_size,
                              hipStream_t stream) {
    const float* x  = (const float*)d_in[0];
    const float* Wq = (const float*)d_in[1];
    const float* bq = (const float*)d_in[2];
    const float* Wk = (const float*)d_in[3];
    const float* bk = (const float*)d_in[4];
    const float* Wv = (const float*)d_in[5];
    const float* bv = (const float*)d_in[6];
    float* out = (float*)d_out;

    const int B = 8, T = 2048, E = 1024, H = 1024;
    const int M = B * T;

    char* ws = (char*)d_ws;
    size_t off = 0;
    bf16_t* xbf = (bf16_t*)(ws + off); off += (size_t)M * E * 2;       // 32 MB
    bf16_t* wqb = (bf16_t*)(ws + off); off += (size_t)E * H * 2;       //  2 MB
    bf16_t* wkb = (bf16_t*)(ws + off); off += (size_t)E * H * 2;
    bf16_t* wvb = (bf16_t*)(ws + off); off += (size_t)E * H * 2;
    bf16_t* qbf = (bf16_t*)(ws + off); off += (size_t)M * H * 2;       // 32 MB
    bf16_t* kbf = (bf16_t*)(ws + off); off += (size_t)M * H * 2;
    bf16_t* vbf = (bf16_t*)(ws + off); off += (size_t)M * H * 2;

    // fp32 -> bf16 conversions
    {
        int n = M * E;
        cvt_bf16_kernel<<<n / (4 * 256), 256, 0, stream>>>(x, xbf, n);
        int nw = E * H;
        cvt_bf16_kernel<<<nw / (4 * 256), 256, 0, stream>>>(Wq, wqb, nw);
        cvt_bf16_kernel<<<nw / (4 * 256), 256, 0, stream>>>(Wk, wkb, nw);
        cvt_bf16_kernel<<<nw / (4 * 256), 256, 0, stream>>>(Wv, wvb, nw);
    }

    // QKV projections
    dim3 ggrid(H / GM_BN, M / GM_BM);
    qkv_gemm_kernel<<<ggrid, 256, 0, stream>>>(xbf, wqb, bq, qbf, M, E, H);
    qkv_gemm_kernel<<<ggrid, 256, 0, stream>>>(xbf, wkb, bk, kbf, M, E, H);
    qkv_gemm_kernel<<<ggrid, 256, 0, stream>>>(xbf, wvb, bv, vbf, M, E, H);

    // causal flash attention
    dim3 agrid(T / AT_BM, B);
    attn_kernel<<<agrid, 256, 0, stream>>>(qbf, kbf, vbf, out, T);
}
